// SplitJointAttention_16758962389591
// MI455X (gfx1250) — matrix operand order repaired and checked by reading, not compiled, not run on hardware
//
#include <hip/hip_runtime.h>
#include <hip/hip_bf16.h>

// ---------------------------------------------------------------------------
// SplitJointAttention for MI455X (gfx1250, wave32, WMMA 16x16x32 f16)
// B=2, T=2048, P=256, DIM=1024, H=16, HD=64, AD=1024
//
// All fragment reads are per-lane contiguous (2 x b128):
//  - weights pre-swizzled to fragment-major layout at f32->f16 conversion
//  - V pre-swizzled to fragment-major layout in the qkv-post kernel
//  - Q/K/P fragment reads done as explicit float4 pairs
// ---------------------------------------------------------------------------

#define Bb   2
#define Tt   2048
#define Pp   256
#define DIM  1024
#define Hh   16
#define HD   64
#define AD   1024
#define EPSF 1e-6f

typedef __attribute__((ext_vector_type(16))) _Float16 v16h;
typedef __attribute__((ext_vector_type(8)))  float    v8f;

union Frag16 {
    v16h      v;
    _Float16  h[16];
    unsigned  u[8];
    float4    q[2];
};

__device__ __forceinline__ v8f wmma_f16(v16h a, v16h b, v8f c) {
    // D = A(16x32 f16) * B(32x16 f16) + C(16x16 f32)
    return __builtin_amdgcn_wmma_f32_16x16x32_f16(
        /*neg_a=*/false, a, /*neg_b=*/false, b,
        /*c_mod=*/(short)0, c, /*reuse_a=*/false, /*reuse_b=*/false);
}

// ---------------------------------------------------------------------------
// f32 -> f16 weight conversion with fragment-major swizzle.
// Element (k, n) of W[K,N] goes to:
//   chunk kc = k/32, tile nt = n/16, lane = (n%16) + 16*((k%32)/16), j = k%16
//   out[((kc*(N/16) + nt)*32 + lane)*16 + j]
// So each (kc, 128-col block) the GEMM needs is one contiguous 8KB region,
// and each lane's B-fragment is 32 contiguous bytes.
// ---------------------------------------------------------------------------
__global__ void cvt_swizzle_w_kernel(const float* __restrict__ in,
                                     _Float16* __restrict__ out,
                                     int K, int N) {
    size_t total = (size_t)K * N;
    size_t i = (size_t)blockIdx.x * blockDim.x + threadIdx.x;
    size_t stride = (size_t)gridDim.x * blockDim.x;
    for (; i < total; i += stride) {
        int n = (int)(i % N);
        int k = (int)(i / N);
        int kc = k >> 5, kr = k & 31;
        int nt = n >> 4, nl = n & 15;
        size_t o = (((size_t)kc * (N >> 4) + nt) * 32 + (nl + 16 * (kr >> 4))) * 16
                 + (kr & 15);
        out[o] = (_Float16)in[i];
    }
}

// ---------------------------------------------------------------------------
// RMSNorm over DIM=1024 -> f16
// ---------------------------------------------------------------------------
__global__ __launch_bounds__(256) void rmsnorm_f16_kernel(
    const float* __restrict__ X, const float* __restrict__ w,
    _Float16* __restrict__ Y) {
    __shared__ float red[8];
    int row = blockIdx.x;
    const float* xr = X + (size_t)row * DIM;
    float v[4];
    float loc = 0.f;
#pragma unroll
    for (int i = 0; i < 4; ++i) {
        v[i] = xr[threadIdx.x + 256 * i];
        loc += v[i] * v[i];
    }
#pragma unroll
    for (int off = 16; off > 0; off >>= 1) loc += __shfl_xor(loc, off, 32);
    if ((threadIdx.x & 31) == 0) red[threadIdx.x >> 5] = loc;
    __syncthreads();
    float tot = 0.f;
#pragma unroll
    for (int i = 0; i < 8; ++i) tot += red[i];
    float scale = rsqrtf(tot / (float)DIM + EPSF);
#pragma unroll
    for (int i = 0; i < 4; ++i) {
        int c = threadIdx.x + 256 * i;
        Y[(size_t)row * DIM + c] = (_Float16)(v[i] * scale * w[c]);
    }
}

// ---------------------------------------------------------------------------
// Tiled WMMA GEMM:  C[M,N] (f32) = A[M,K] (f16, row-major) * Wsw (f16,
// fragment-swizzled) + bias.  Block tile 64x128, K-step 32, 8 waves,
// each wave -> 16x64 (4 x 16x16 acc).  Optional per-row uint8 mask.
// ---------------------------------------------------------------------------
__global__ __launch_bounds__(256) void gemm_wmma_kernel(
    const _Float16* __restrict__ A, const _Float16* __restrict__ Wsw,
    const float* __restrict__ bias, const unsigned char* __restrict__ rowMask,
    float* __restrict__ C, int M, int K, int N) {
    __shared__ _Float16 As[64][32];
    __shared__ _Float16 Bs[8][32][16];   // [n-tile][lane][j]

    int tid  = threadIdx.x;
    int wave = tid >> 5;
    int lane = tid & 31;
    int n16  = lane & 15;
    int hi   = lane >> 4;

    int mBase = blockIdx.x * 64;
    int nBase = blockIdx.y * 128;
    int mOff  = (wave & 3) * 16;      // wave's 16-row subtile
    int nGrp  = (wave >> 2) * 64;     // wave's 64-col group (4 subtiles)
    int ntBase = nGrp >> 4;           // 0 or 4

    v8f acc[4];
    v8f zero = {0.f, 0.f, 0.f, 0.f, 0.f, 0.f, 0.f, 0.f};
#pragma unroll
    for (int i = 0; i < 4; ++i) acc[i] = zero;

    int aRow = tid >> 2;          // 0..63
    int aCol = (tid & 3) * 8;     // 0..24
    int nTiles = N >> 4;

    for (int k0 = 0; k0 < K; k0 += 32) {
        __syncthreads();
        // A tile: 64x32 halves, one float4 per thread (coalesced)
        *(float4*)&As[aRow][aCol] =
            *(const float4*)(A + (size_t)(mBase + aRow) * K + k0 + aCol);
        // B tile: contiguous 4096-half region (pre-swizzled), 2 float4/thread
        const float4* bsrc = (const float4*)(
            Wsw + (((size_t)(k0 >> 5)) * nTiles + (nBase >> 4)) * 512);
        float4* bdst = (float4*)&Bs[0][0][0];
        bdst[2 * tid]     = bsrc[2 * tid];
        bdst[2 * tid + 1] = bsrc[2 * tid + 1];
        if (k0 + 32 < K)
            __builtin_prefetch(A + (size_t)(mBase + aRow) * K + k0 + 32 + aCol, 0, 1);
        __syncthreads();

        // A fragment: two contiguous b128 per lane
        Frag16 af;
        af.q[0] = *(const float4*)&As[mOff + n16][16 * hi];
        af.q[1] = *(const float4*)&As[mOff + n16][16 * hi + 8];
#pragma unroll
        for (int nt = 0; nt < 4; ++nt) {
            Frag16 bf;  // fragment-major: lane's 32 bytes contiguous
            const float4* bp = (const float4*)&Bs[ntBase + nt][lane][0];
            bf.q[0] = bp[0];
            bf.q[1] = bp[1];
            acc[nt] = wmma_f16(af.v, bf.v, acc[nt]);
        }
    }

#pragma unroll
    for (int nt = 0; nt < 4; ++nt) {
#pragma unroll
        for (int r = 0; r < 8; ++r) {
            int row = mBase + mOff + r + 8 * hi;
            int col = nBase + nGrp + nt * 16 + n16;
            float v = acc[nt][r];
            if (bias) v += bias[col];
            if (rowMask) v *= (float)rowMask[row];
            C[(size_t)row * N + col] = v;
        }
    }
}

// ---------------------------------------------------------------------------
// Per-head RMSNorm(q,k) + RoPE.  Q,K written row-major (B,H,L,64).
// V written fragment-swizzled per (b,h): for key chunk kc=l/32:
//   [ (bh*(L/32)+kc)*4 + nt ]*512 + lane*16 + j
//   with lane = (d%16) + 16*((l%32)/16), j = l%16, nt = d/16
// One block of 64 threads per (b, l, h). grid.x = B*L*H
// ---------------------------------------------------------------------------
__global__ __launch_bounds__(64) void qkv_post_kernel(
    const float* __restrict__ QKV,   // (B, L, 3*AD)
    const float* __restrict__ wq, const float* __restrict__ wk,  // (64)
    _Float16* __restrict__ Qh, _Float16* __restrict__ Kh,
    _Float16* __restrict__ Vh, int L) {
    __shared__ float redq[2], redk[2];
    __shared__ float qn[HD], kn[HD];

    int d = threadIdx.x;
    int h = blockIdx.x % Hh;
    int l = (blockIdx.x / Hh) % L;
    int b = blockIdx.x / (Hh * L);
    int bh = b * Hh + h;

    const float* base = QKV + (size_t)(b * L + l) * (3 * AD);
    float q  = base[h * HD + d];
    float k  = base[AD + h * HD + d];
    float vv = base[2 * AD + h * HD + d];

    float sq = q * q, sk = k * k;
#pragma unroll
    for (int off = 16; off > 0; off >>= 1) {
        sq += __shfl_xor(sq, off, 32);
        sk += __shfl_xor(sk, off, 32);
    }
    if ((d & 31) == 0) { redq[d >> 5] = sq; redk[d >> 5] = sk; }
    __syncthreads();
    float msq = (redq[0] + redq[1]) / (float)HD;
    float msk = (redk[0] + redk[1]) / (float)HD;
    qn[d] = q * rsqrtf(msq + EPSF) * wq[d];
    kn[d] = k * rsqrtf(msk + EPSF) * wk[d];
    __syncthreads();

    int i = d & 31;  // rotary pair index (half = 32)
    float invf = powf(10000.f, -(float)i / 32.f);
    float ang  = (float)l * invf;
    float c = cosf(ang), s = sinf(ang);
    float qo, ko;
    if (d < 32) { qo = qn[d] * c - qn[d + 32] * s;  ko = kn[d] * c - kn[d + 32] * s; }
    else        { qo = qn[d - 32] * s + qn[d] * c;  ko = kn[d - 32] * s + kn[d] * c; }

    size_t oidx = (((size_t)bh * L) + l) * HD + d;
    Qh[oidx] = (_Float16)qo;
    Kh[oidx] = (_Float16)ko;

    // V fragment-swizzled store
    int kc = l >> 5, kr = l & 31;
    int vlane = (d & 15) + 16 * (kr >> 4);
    int vj = kr & 15;
    int nt = d >> 4;
    size_t vidx = (((size_t)bh * (L >> 5) + kc) * 4 + nt) * 512
                + (size_t)vlane * 16 + vj;
    Vh[vidx] = (_Float16)vv;
}

// ---------------------------------------------------------------------------
// Flash attention, f16 WMMA.  Block = 8 waves = 128 q rows; key tiles of 32.
// Q,K row-major (B,H,L,64) f16; V fragment-swizzled.  kmask: (B,Lk) uint8.
// Output f16 into concat buffer (B, Lq, 2*AD) at column colOff + h*64.
// ---------------------------------------------------------------------------
__global__ __launch_bounds__(256) void flash_attn_kernel(
    const _Float16* __restrict__ Qh, const _Float16* __restrict__ Kh,
    const _Float16* __restrict__ Vh, const unsigned char* __restrict__ kmask,
    _Float16* __restrict__ Out, int colOff, int Lq, int Lk) {
    __shared__ _Float16 Ks[32][HD];          // row-major key tile
    __shared__ _Float16 Vs[4][32][16];       // fragment-major V tile
    __shared__ _Float16 Ps[8][16][32];       // per-wave P staging

    int tid = threadIdx.x, wave = tid >> 5, lane = tid & 31;
    int n16 = lane & 15, hi = lane >> 4;
    int bh = blockIdx.y;
    int b = bh >> 4, h = bh & 15;
    int qBase = blockIdx.x * 128 + wave * 16;

    const _Float16* Qp = Qh + (size_t)bh * Lq * HD;
    const _Float16* Kp = Kh + (size_t)bh * Lk * HD;
    const _Float16* Vp = Vh + (size_t)bh * Lk * HD;  // swizzled, same extent

    // Preload Q fragments (two 32-wide K-dim chunks), contiguous b128 pairs
    Frag16 qa[2];
    {
        const _Float16* qrow = Qp + (size_t)(qBase + n16) * HD;
#pragma unroll
        for (int c = 0; c < 2; ++c) {
            qa[c].q[0] = *(const float4*)(qrow + 32 * c + 16 * hi);
            qa[c].q[1] = *(const float4*)(qrow + 32 * c + 16 * hi + 8);
        }
    }

    v8f zero = {0.f, 0.f, 0.f, 0.f, 0.f, 0.f, 0.f, 0.f};
    v8f O[4];
#pragma unroll
    for (int i = 0; i < 4; ++i) O[i] = zero;
    float rm[8], rl[8];
#pragma unroll
    for (int r = 0; r < 8; ++r) { rm[r] = -1e30f; rl[r] = 0.f; }

    int ldRow = tid >> 3;          // 0..31
    int ldCol = (tid & 7) * 8;     // 0..56

    for (int kt = 0; kt < Lk; kt += 32) {
        __syncthreads();
        // K tile: row-major, one float4 per thread
        *(float4*)&Ks[ldRow][ldCol] =
            *(const float4*)(Kp + (size_t)(kt + ldRow) * HD + ldCol);
        // V tile: contiguous swizzled 2048-half region, one float4 per thread
        ((float4*)&Vs[0][0][0])[tid] =
            ((const float4*)(Vp + (size_t)(kt >> 5) * 2048))[tid];
        __syncthreads();

        // S = Q * K^T -> two 16x16 f32 tiles (keys sub*16 .. sub*16+15)
        v8f S[2];
        S[0] = zero; S[1] = zero;
#pragma unroll
        for (int sub = 0; sub < 2; ++sub) {
#pragma unroll
            for (int c = 0; c < 2; ++c) {
                Frag16 bf;  // B[k][n] = K[key sub*16+n][32c + 16hi + j], contiguous
                const float4* ksrc =
                    (const float4*)&Ks[sub * 16 + n16][32 * c + 16 * hi];
                bf.q[0] = ksrc[0];
                bf.q[1] = ksrc[1];
                S[sub] = wmma_f16(qa[c].v, bf.v, S[sub]);
            }
        }

        // key-mask per lane column
        float mk0 = kmask[(size_t)b * Lk + kt + n16]      ? 0.f : -1e9f;
        float mk1 = kmask[(size_t)b * Lk + kt + 16 + n16] ? 0.f : -1e9f;

        // online softmax per row (each 16-lane group owns 8 rows)
#pragma unroll
        for (int r = 0; r < 8; ++r) {
            float v0 = S[0][r] * 0.125f + mk0;
            float v1 = S[1][r] * 0.125f + mk1;
            float mx = fmaxf(v0, v1);
#pragma unroll
            for (int off = 8; off > 0; off >>= 1)
                mx = fmaxf(mx, __shfl_xor(mx, off, 32));
            float mnew = fmaxf(rm[r], mx);
            float a  = __expf(rm[r] - mnew);
            float p0 = __expf(v0 - mnew);
            float p1 = __expf(v1 - mnew);
            float ps = p0 + p1;
#pragma unroll
            for (int off = 8; off > 0; off >>= 1)
                ps += __shfl_xor(ps, off, 32);
            rl[r] = rl[r] * a + ps;
            rm[r] = mnew;
#pragma unroll
            for (int nt = 0; nt < 4; ++nt) O[nt][r] *= a;
            Ps[wave][r + 8 * hi][n16]      = (_Float16)p0;
            Ps[wave][r + 8 * hi][16 + n16] = (_Float16)p1;
        }
        __syncthreads();  // publish P tiles (uniform across waves)

        // O += P(16x32) * V(32x64)
        Frag16 pa;
        pa.q[0] = *(const float4*)&Ps[wave][n16][16 * hi];
        pa.q[1] = *(const float4*)&Ps[wave][n16][16 * hi + 8];
#pragma unroll
        for (int nt = 0; nt < 4; ++nt) {
            Frag16 vb;  // fragment-major: lane's 32 bytes contiguous
            const float4* vsrc = (const float4*)&Vs[nt][lane][0];
            vb.q[0] = vsrc[0];
            vb.q[1] = vsrc[1];
            O[nt] = wmma_f16(pa.v, vb.v, O[nt]);
        }
    }

    // epilogue: normalize and scatter to concat buffer
#pragma unroll
    for (int r = 0; r < 8; ++r) {
        float invl = 1.f / rl[r];
        int qr = qBase + r + 8 * hi;
#pragma unroll
        for (int nt = 0; nt < 4; ++nt) {
            int col = colOff + h * HD + nt * 16 + n16;
            Out[(size_t)(b * Lq + qr) * (2 * AD) + col] = (_Float16)(O[nt][r] * invl);
        }
    }
}

// ---------------------------------------------------------------------------
// Host-side launch
// ---------------------------------------------------------------------------
extern "C" void kernel_launch(void* const* d_in, const int* in_sizes, int n_in,
                              void* d_out, int out_size, void* d_ws, size_t ws_size,
                              hipStream_t stream) {
    (void)in_sizes; (void)n_in; (void)out_size; (void)ws_size;

    const float* pool        = (const float*)d_in[0];
    const float* x           = (const float*)d_in[1];
    const unsigned char* t_mask = (const unsigned char*)d_in[3];
    const unsigned char* n_mask = (const unsigned char*)d_in[4];
    const float* w_pool_norm = (const float*)d_in[9];
    const float* w_x_norm    = (const float*)d_in[10];
    const float* w_pool_qkv  = (const float*)d_in[11];
    const float* b_pool_qkv  = (const float*)d_in[12];
    const float* w_x_qkv     = (const float*)d_in[13];
    const float* b_x_qkv     = (const float*)d_in[14];
    const float* w_pq_norm   = (const float*)d_in[15];
    const float* w_pk_norm   = (const float*)d_in[16];
    const float* w_xq_norm   = (const float*)d_in[17];
    const float* w_xk_norm   = (const float*)d_in[18];
    const float* w_pool_merge= (const float*)d_in[19];
    const float* b_pool_merge= (const float*)d_in[20];
    const float* w_x_merge   = (const float*)d_in[21];
    const float* b_x_merge   = (const float*)d_in[22];

    float* out_pool = (float*)d_out;                          // (B,P,DIM)
    float* out_x    = (float*)d_out + (size_t)Bb * Pp * DIM;  // (B,T,DIM)

    // workspace carve-up
    char* ws = (char*)d_ws;
    size_t off = 0;
    auto alloc = [&](size_t bytes) -> char* {
        char* p = ws + off;
        off += (bytes + 255) & ~(size_t)255;
        return p;
    };
    const size_t Mp = (size_t)Bb * Pp;   // 512
    const size_t Mx = (size_t)Bb * Tt;   // 4096

    _Float16* poolN  = (_Float16*)alloc(Mp * DIM * 2);
    _Float16* xN     = (_Float16*)alloc(Mx * DIM * 2);
    _Float16* wPQKVh = (_Float16*)alloc((size_t)DIM * 3 * AD * 2);
    _Float16* wXQKVh = (_Float16*)alloc((size_t)DIM * 3 * AD * 2);
    _Float16* wPMh   = (_Float16*)alloc((size_t)2 * AD * DIM * 2);
    _Float16* wXMh   = (_Float16*)alloc((size_t)2 * AD * DIM * 2);
    float*    poolQkv= (float*)alloc(Mp * 3 * AD * 4);
    float*    xQkv   = (float*)alloc(Mx * 3 * AD * 4);
    _Float16* pq     = (_Float16*)alloc(Mp * AD * 2);
    _Float16* pk     = (_Float16*)alloc(Mp * AD * 2);
    _Float16* pv     = (_Float16*)alloc(Mp * AD * 2);
    _Float16* xq     = (_Float16*)alloc(Mx * AD * 2);
    _Float16* xk     = (_Float16*)alloc(Mx * AD * 2);
    _Float16* xv     = (_Float16*)alloc(Mx * AD * 2);
    _Float16* poolCat= (_Float16*)alloc(Mp * 2 * AD * 2);
    _Float16* xCat   = (_Float16*)alloc(Mx * 2 * AD * 2);

    // 1) weights -> f16, fragment-major swizzle
    cvt_swizzle_w_kernel<<<2048, 256, 0, stream>>>(w_pool_qkv,   wPQKVh, DIM,    3 * AD);
    cvt_swizzle_w_kernel<<<2048, 256, 0, stream>>>(w_x_qkv,      wXQKVh, DIM,    3 * AD);
    cvt_swizzle_w_kernel<<<2048, 256, 0, stream>>>(w_pool_merge, wPMh,   2 * AD, DIM);
    cvt_swizzle_w_kernel<<<2048, 256, 0, stream>>>(w_x_merge,    wXMh,   2 * AD, DIM);

    // 2) RMSNorm -> f16
    rmsnorm_f16_kernel<<<(int)Mp, 256, 0, stream>>>(pool, w_pool_norm, poolN);
    rmsnorm_f16_kernel<<<(int)Mx, 256, 0, stream>>>(x,    w_x_norm,    xN);

    // 3) QKV GEMMs: (M x 1024) * (1024 x 3072)
    gemm_wmma_kernel<<<dim3(Mp / 64, 3 * AD / 128), 256, 0, stream>>>(
        poolN, wPQKVh, b_pool_qkv, nullptr, poolQkv, (int)Mp, DIM, 3 * AD);
    gemm_wmma_kernel<<<dim3(Mx / 64, 3 * AD / 128), 256, 0, stream>>>(
        xN, wXQKVh, b_x_qkv, nullptr, xQkv, (int)Mx, DIM, 3 * AD);

    // 4) per-head RMS + RoPE -> Q,K row-major; V fragment-swizzled
    qkv_post_kernel<<<Bb * Pp * Hh, 64, 0, stream>>>(
        poolQkv, w_pq_norm, w_pk_norm, pq, pk, pv, Pp);
    qkv_post_kernel<<<Bb * Tt * Hh, 64, 0, stream>>>(
        xQkv, w_xq_norm, w_xk_norm, xq, xk, xv, Tt);

    // 5) four attentions (key-only masks, per reference)
    flash_attn_kernel<<<dim3(Pp / 128, Bb * Hh), 256, 0, stream>>>(
        pq, pk, pv, n_mask, poolCat, 0,  Pp, Pp);       // pp
    flash_attn_kernel<<<dim3(Pp / 128, Bb * Hh), 256, 0, stream>>>(
        pq, xk, xv, t_mask, poolCat, AD, Pp, Tt);       // px
    flash_attn_kernel<<<dim3(Tt / 128, Bb * Hh), 256, 0, stream>>>(
        xq, xk, xv, t_mask, xCat, 0,  Tt, Tt);          // xx
    flash_attn_kernel<<<dim3(Tt / 128, Bb * Hh), 256, 0, stream>>>(
        xq, pk, pv, n_mask, xCat, AD, Tt, Pp);          // xp

    // 6) merge GEMMs: (M x 2048) * (2048 x 1024) + bias, row-masked -> d_out
    gemm_wmma_kernel<<<dim3(Mp / 64, DIM / 128), 256, 0, stream>>>(
        poolCat, wPMh, b_pool_merge, n_mask, out_pool, (int)Mp, 2 * AD, DIM);
    gemm_wmma_kernel<<<dim3(Mx / 64, DIM / 128), 256, 0, stream>>>(
        xCat, wXMh, b_x_merge, t_mask, out_x, (int)Mx, 2 * AD, DIM);
}
